// TransBlock_33036888441084
// MI455X (gfx1250) — matrix operand order repaired and checked by reading, not compiled, not run on hardware
//
#include <hip/hip_runtime.h>
#include <math.h>

// ---------------------------------------------------------------------------
// Types for CDNA5 WMMA (wave32, 16x16x32 bf16 -> f32)
// ---------------------------------------------------------------------------
typedef __attribute__((ext_vector_type(16))) __bf16 v16bf;
typedef __attribute__((ext_vector_type(8)))  float  v8f;
typedef __attribute__((ext_vector_type(4)))  unsigned int v4u;
typedef __attribute__((ext_vector_type(8)))  unsigned int v8u;

#define NN 5000       // nodes
#define EE 50000      // edges
#define SS 9          // (L+1)^2
#define CC 128        // channels
#define ROW 1152      // SS*CC
#define KSTEP 32      // k-slice per TDM stage

__device__ __forceinline__ unsigned short f2bf(float f) {
  unsigned u = __float_as_uint(f);
  unsigned r = u + 0x7FFFu + ((u >> 16) & 1u);   // round-to-nearest-even
  return (unsigned short)(r >> 16);
}
__device__ __forceinline__ float bf2f(unsigned short u) {
  return __uint_as_float(((unsigned)u) << 16);
}
__device__ __forceinline__ float siluf(float x) { return x / (1.f + expf(-x)); }

// monotone float<->uint key for atomicMax-based segment max
__device__ __forceinline__ unsigned enc_f(float f) {
  unsigned u = __float_as_uint(f);
  return (u & 0x80000000u) ? ~u : (u | 0x80000000u);
}
__device__ __forceinline__ float dec_f(unsigned k) {
  unsigned u = (k & 0x80000000u) ? (k & 0x7FFFFFFFu) : ~k;
  return __uint_as_float(u);
}

// LDS byte offset of a shared-memory object (flat LDS aperture keeps the
// offset in the low 32 bits)
__device__ __forceinline__ unsigned lds_off_of(const void* p) {
  return (unsigned)(unsigned long long)p;
}

// ---------------------------------------------------------------------------
// TDM: issue a 2D bf16 tile load Global->LDS (one per wave; EXEC ignored).
// D# group0/group1 packed per CDNA5 ISA 8.3/8.4. 2D => VADDR2/3 omitted.
//   tdim0/tdim1 : remaining tensor extent from tile start (OOB reads -> 0)
//   tile0/tile1 : tile dims (elements / rows)
//   stride0     : elements between consecutive rows
// ---------------------------------------------------------------------------
__device__ __forceinline__ void tdm_load_2d(unsigned lds_addr, const void* gptr,
                                            unsigned tdim0, unsigned tdim1,
                                            unsigned tile0, unsigned tile1,
                                            unsigned stride0) {
  unsigned long long ga = (unsigned long long)gptr;
  v4u g0;
  g0[0] = 1u;                                        // count=1, no gather
  g0[1] = lds_addr;                                  // lds_addr [63:32]
  g0[2] = (unsigned)ga;                              // global_addr [95:64]
  g0[3] = (unsigned)((ga >> 32) & 0x01FFFFFFu) | (2u << 30);  // [120:96] | type=2
  v8u g1;
  g1[0] = 1u << 16;                                  // wg_mask=0, data_size=1 (2B)
  g1[1] = (tdim0 & 0xFFFFu) << 16;                   // tensor_dim0 [15:0]
  g1[2] = (tdim0 >> 16) | ((tdim1 & 0xFFFFu) << 16); // dim0 hi | tensor_dim1 lo
  g1[3] = (tdim1 >> 16) | (tile0 << 16);             // dim1 hi | tile_dim0
  g1[4] = tile1;                                     // tile_dim1 (tile_dim2=0)
  g1[5] = stride0;                                   // tensor_dim0_stride lo
  g1[6] = 0u;                                        // stride0 hi | dim1_stride lo
  g1[7] = 0u;
  asm volatile("tensor_load_to_lds %0, %1" :: "s"(g0), "s"(g1) : "memory");
}

// ---------------------------------------------------------------------------
// K0: transpose f32 [K,N] weight -> bf16 [N,K] (B operand for WMMA)
// ---------------------------------------------------------------------------
__global__ void convT_bf16(const float* __restrict__ in, unsigned short* __restrict__ out,
                           int K, int N) {
  int idx = blockIdx.x * 256 + threadIdx.x;
  if (idx >= K * N) return;
  int n = idx / K, k = idx - n * K;
  out[idx] = f2bf(in[(size_t)k * N + n]);
}

// ---------------------------------------------------------------------------
// K1/K11: equivariant norm.  block=128 threads (=channel), grid=NN
// ---------------------------------------------------------------------------
__global__ void eqv_norm_kernel(const float* __restrict__ x, const float* __restrict__ g,
                                const float* __restrict__ b, float* __restrict__ out) {
  int n = blockIdx.x, c = threadIdx.x;
  __shared__ float red[128];
  float v[SS];
#pragma unroll
  for (int s = 0; s < SS; ++s) v[s] = x[(size_t)n * ROW + s * CC + c];
  // l = 0 : layernorm over channels
  red[c] = v[0]; __syncthreads();
  for (int o = 64; o > 0; o >>= 1) { if (c < o) red[c] += red[c + o]; __syncthreads(); }
  float mu = red[0] / 128.f; __syncthreads();
  float d0 = v[0] - mu;
  red[c] = d0 * d0; __syncthreads();
  for (int o = 64; o > 0; o >>= 1) { if (c < o) red[c] += red[c + o]; __syncthreads(); }
  float var = red[0] / 128.f; __syncthreads();
  out[(size_t)n * ROW + c] = d0 * rsqrtf(var + 1e-5f) * g[c] + b[c];
  // l = 1 : rows 1..3 RMS over (3,128)
  float ss = v[1]*v[1] + v[2]*v[2] + v[3]*v[3];
  red[c] = ss; __syncthreads();
  for (int o = 64; o > 0; o >>= 1) { if (c < o) red[c] += red[c + o]; __syncthreads(); }
  float rms1 = 1.f / sqrtf(red[0] / (3.f * 128.f) + 1e-5f);
  __syncthreads();
#pragma unroll
  for (int s = 1; s <= 3; ++s)
    out[(size_t)n * ROW + s * CC + c] = v[s] * rms1 * g[CC + c];
  // l = 2 : rows 4..8 RMS over (5,128)
  float s2 = v[4]*v[4] + v[5]*v[5] + v[6]*v[6] + v[7]*v[7] + v[8]*v[8];
  red[c] = s2; __syncthreads();
  for (int o = 64; o > 0; o >>= 1) { if (c < o) red[c] += red[c + o]; __syncthreads(); }
  float rms2 = 1.f / sqrtf(red[0] / (5.f * 128.f) + 1e-5f);
  __syncthreads();
#pragma unroll
  for (int s = 4; s <= 8; ++s)
    out[(size_t)n * ROW + s * CC + c] = v[s] * rms2 * g[2 * CC + c];
}

// ---------------------------------------------------------------------------
// K2: radial MLP  rad[e,768] = silu(LN(ed@W1+b1)) @ W2 + b2 . grid=EE, block=256
// ---------------------------------------------------------------------------
__global__ void radial_kernel(const float* __restrict__ ed, const float* __restrict__ w1,
                              const float* __restrict__ b1, const float* __restrict__ lng,
                              const float* __restrict__ lnb, const float* __restrict__ w2,
                              const float* __restrict__ b2, float* __restrict__ rad) {
  int e = blockIdx.x, t = threadIdx.x;
  __shared__ float edv[128], h[64], red[64];
  if (t < 128) edv[t] = ed[(size_t)e * 128 + t];
  __syncthreads();
  if (t < 64) {
    float s = 0.f;
    for (int k = 0; k < 128; ++k) s += edv[k] * w1[k * 64 + t];
    h[t] = s + b1[t];
  }
  __syncthreads();
  if (t < 64) red[t] = h[t];
  __syncthreads();
  for (int o = 32; o > 0; o >>= 1) { if (t < o) red[t] += red[t + o]; __syncthreads(); }
  float mu = red[0] / 64.f; __syncthreads();
  if (t < 64) { float d = h[t] - mu; red[t] = d * d; }
  __syncthreads();
  for (int o = 32; o > 0; o >>= 1) { if (t < o) red[t] += red[t + o]; __syncthreads(); }
  float var = red[0] / 64.f; __syncthreads();
  if (t < 64) {
    float d = (h[t] - mu) * rsqrtf(var + 1e-5f) * lng[t] + lnb[t];
    h[t] = siluf(d);
  }
  __syncthreads();
  for (int o = t; o < 768; o += 256) {
    float s = 0.f;
    for (int k = 0; k < 64; ++k) s += h[k] * w2[k * 768 + o];
    rad[(size_t)e * 768 + o] = s + b2[o];
  }
}

// ---------------------------------------------------------------------------
// K3: gather + wigner rotate + radial scale -> bf16 A matrix, rows in m-order
// PERM = [0,2,6,3,7,1,5,8,4] (m-row p holds l-major row PERM[p])
// grid=EE, block=256 (thread = concat channel)
// ---------------------------------------------------------------------------
__global__ void build_msg_kernel(const float* __restrict__ xn, const float* __restrict__ wig,
                                 const float* __restrict__ rad, const int* __restrict__ esrc,
                                 const int* __restrict__ edst, unsigned short* __restrict__ msgm) {
  int e = blockIdx.x, t = threadIdx.x;
  __shared__ float W[81];
  if (t < 81) W[t] = wig[(size_t)e * 81 + t];
  int src = esrc[e], dst = edst[e];
  const float* base = (t < 128) ? (xn + (size_t)src * ROW + t)
                                : (xn + (size_t)dst * ROW + (t - 128));
  float v[SS];
#pragma unroll
  for (int j = 0; j < SS; ++j) v[j] = base[j * CC];
  __syncthreads();
  const float* re = rad + (size_t)e * 768;
  const int PERM[SS] = {0, 2, 6, 3, 7, 1, 5, 8, 4};
  const int LOF[SS]  = {0, 1, 1, 1, 2, 2, 2, 2, 2};
#pragma unroll
  for (int p = 0; p < SS; ++p) {
    int i = PERM[p];
    float s = 0.f;
#pragma unroll
    for (int j = 0; j < SS; ++j) s += W[i * 9 + j] * v[j];
    s *= re[LOF[i] * 256 + t];
    msgm[(size_t)e * 2304 + p * 256 + t] = f2bf(s);
  }
}

// ---------------------------------------------------------------------------
// WMMA GEMM with TDM double-buffered LDS staging.
// C[M,N] = A[M,K](bf16, row stride lda) x BT[N,K](bf16) (+bias), f32 out.
// block = 256 (8 waves). Block tile 128(M) x 64(N); wave w owns rows
// [w*16, w*16+16). Per stage: wave0 issues tensor_load_to_lds for the NEXT
// A(128xKSTEP) and B(64xKSTEP) slices, waits TENSORcnt<=2 (in-order => current
// stage resident), barrier, all waves ds_load fragments + 4x v_wmma.
// M-tail rows are zero-filled by TDM OOB handling; stores are guarded.
// ---------------------------------------------------------------------------
__global__ void gemm_bf16_wmma(const unsigned short* __restrict__ A, int lda,
                               const unsigned short* __restrict__ BT,
                               const float* __restrict__ bias,
                               float* __restrict__ C, int M, int N, int K) {
  __shared__ unsigned short tA[2][128 * KSTEP];
  __shared__ unsigned short tB[2][64 * KSTEP];
  const int lane = threadIdx.x & 31;
  const int wave = threadIdx.x >> 5;
  const int t  = lane & 15;
  const int hi = lane >> 4;
  const int row0  = blockIdx.y * 128;
  const int nbase = blockIdx.x * 64;
  const int nst = K / KSTEP;

  v8f acc[4];
#pragma unroll
  for (int j = 0; j < 4; ++j) acc[j] = (v8f){0.f,0.f,0.f,0.f,0.f,0.f,0.f,0.f};

  if (wave == 0) {
    tdm_load_2d(lds_off_of(tA[0]), A + (size_t)row0 * lda,
                (unsigned)K, (unsigned)(M - row0), KSTEP, 128, (unsigned)lda);
    tdm_load_2d(lds_off_of(tB[0]), BT + (size_t)nbase * K,
                (unsigned)K, (unsigned)(N - nbase), KSTEP, 64, (unsigned)K);
  }
  for (int i = 0; i < nst; ++i) {
    if (wave == 0) {
      if (i + 1 < nst) {
        int kb = (i + 1) * KSTEP;
        tdm_load_2d(lds_off_of(tA[(i + 1) & 1]), A + (size_t)row0 * lda + kb,
                    (unsigned)(K - kb), (unsigned)(M - row0), KSTEP, 128, (unsigned)lda);
        tdm_load_2d(lds_off_of(tB[(i + 1) & 1]), BT + (size_t)nbase * K + kb,
                    (unsigned)(K - kb), (unsigned)(N - nbase), KSTEP, 64, (unsigned)K);
        __builtin_amdgcn_s_wait_tensorcnt(2);   // stage i resident (in-order)
      } else {
        __builtin_amdgcn_s_wait_tensorcnt(0);
      }
    }
    __syncthreads();
    // A fragment from LDS: lane half hi -> K = 8*hi..+7 and 16+8*hi..+7
    const unsigned short* aL = &tA[i & 1][(wave * 16 + t) * KSTEP];
    v16bf af;
    *((v4u*)&af)     = *(const v4u*)(aL + hi * 8);
    *((v4u*)&af + 1) = *(const v4u*)(aL + hi * 8 + 16);
#pragma unroll
    for (int j = 0; j < 4; ++j) {
      // B fragment: lane half hi -> K = 8*hi..+7 and 16+8*hi..+7
      const unsigned short* bL = &tB[i & 1][(j * 16 + t) * KSTEP];
      v16bf bf;
      *((v4u*)&bf)     = *(const v4u*)(bL + hi * 8);
      *((v4u*)&bf + 1) = *(const v4u*)(bL + hi * 8 + 16);
      acc[j] = __builtin_amdgcn_wmma_f32_16x16x32_bf16(
          false, af, false, bf, (short)0, acc[j], false, false);
    }
    __syncthreads();
  }
  const int mbase = row0 + wave * 16;
#pragma unroll
  for (int j = 0; j < 4; ++j) {
    int n = nbase + j * 16 + t;
    float bv = bias ? bias[n] : 0.f;
#pragma unroll
    for (int r = 0; r < 8; ++r) {
      int m = mbase + r + 8 * hi;
      if (m < M) C[(size_t)m * N + n] = acc[j][r] + bv;
    }
  }
}

// ---------------------------------------------------------------------------
// K5: combine conv1 outputs, separable S2 activation, emit bf16 A for conv2
// grid=EE, block=128 (thread = hidden channel)
// ---------------------------------------------------------------------------
__global__ void mid_kernel(const float* __restrict__ Y0, const float* __restrict__ YR1,
                           const float* __restrict__ YI1, const float* __restrict__ YR2,
                           const float* __restrict__ YI2, const float* __restrict__ togrid,
                           const float* __restrict__ fromgrid, unsigned short* __restrict__ hid2m) {
  int e = blockIdx.x, c = threadIdx.x;
  __shared__ float tg[162], fg[162];
  for (int i = c; i < 162; i += 128) { tg[i] = togrid[i]; fg[i] = fromgrid[i]; }
  __syncthreads();
  const float* y0  = Y0  + (size_t)e * 768;
  const float* yr1 = YR1 + (size_t)e * 512;
  const float* yi1 = YI1 + (size_t)e * 512;
  const float* yr2 = YR2 + (size_t)e * 256;
  const float* yi2 = YI2 + (size_t)e * 256;
  float hidl[SS];
  hidl[0] = y0[c]; hidl[2] = y0[128 + c]; hidl[6] = y0[256 + c];
  hidl[3] = yr1[c]       - yi1[256 + c];
  hidl[7] = yr1[128 + c] - yi1[384 + c];
  hidl[1] = yr1[256 + c] + yi1[c];
  hidl[5] = yr1[384 + c] + yi1[128 + c];
  hidl[8] = yr2[c]       - yi2[128 + c];
  hidl[4] = yr2[128 + c] + yi2[c];
  float tacc[SS];
#pragma unroll
  for (int s = 0; s < SS; ++s) tacc[s] = 0.f;
  for (int gi = 0; gi < 18; ++gi) {
    float gv = 0.f;
#pragma unroll
    for (int s = 0; s < SS; ++s) gv += tg[gi * 9 + s] * hidl[s];
    float sg = siluf(gv);
#pragma unroll
    for (int s = 0; s < SS; ++s) tacc[s] += fg[s * 18 + gi] * sg;
  }
  float outl[SS];
  outl[0] = siluf(y0[640 + c]);            // gating channels = y0[640:768]
#pragma unroll
  for (int s = 1; s < SS; ++s) outl[s] = tacc[s];
  const int PERM[SS] = {0, 2, 6, 3, 7, 1, 5, 8, 4};
#pragma unroll
  for (int p = 0; p < SS; ++p)
    hid2m[(size_t)e * ROW + p * CC + c] = f2bf(outl[PERM[p]]);
}

// ---------------------------------------------------------------------------
// K7: combine conv2 outputs, wigner^T rotate -> val_rot (bf16), alpha head
// grid=EE, block=256
// ---------------------------------------------------------------------------
__global__ void post_kernel(const float* __restrict__ Z0, const float* __restrict__ ZR1,
                            const float* __restrict__ ZI1, const float* __restrict__ ZR2,
                            const float* __restrict__ ZI2, const float* __restrict__ Y0,
                            const float* __restrict__ wig, const float* __restrict__ alng,
                            const float* __restrict__ alnb, const float* __restrict__ adot,
                            unsigned short* __restrict__ valrot, float* __restrict__ aout) {
  int e = blockIdx.x, t = threadIdx.x;
  __shared__ float W[81];
  __shared__ float val[SS][CC];
  __shared__ float red[256];
  for (int i = t; i < 81; i += 256) W[i] = wig[(size_t)e * 81 + i];
  if (t < 128) {
    int c = t;
    const float* z0  = Z0  + (size_t)e * 384;
    const float* zr1 = ZR1 + (size_t)e * 512;
    const float* zi1 = ZI1 + (size_t)e * 512;
    const float* zr2 = ZR2 + (size_t)e * 256;
    const float* zi2 = ZI2 + (size_t)e * 256;
    val[0][c] = z0[c]; val[2][c] = z0[128 + c]; val[6][c] = z0[256 + c];
    val[3][c] = zr1[c]       - zi1[256 + c];
    val[7][c] = zr1[128 + c] - zi1[384 + c];
    val[1][c] = zr1[256 + c] + zi1[c];
    val[5][c] = zr1[384 + c] + zi1[128 + c];
    val[8][c] = zr2[c]       - zi2[128 + c];
    val[4][c] = zr2[128 + c] + zi2[c];
  }
  __syncthreads();
  if (t < 128) {
    int c = t;
#pragma unroll
    for (int i = 0; i < SS; ++i) {
      float s = 0.f;
#pragma unroll
      for (int j = 0; j < SS; ++j) s += W[j * 9 + i] * val[j][c];   // wigner^T
      valrot[(size_t)e * ROW + i * CC + c] = f2bf(s);
    }
  }
  // alpha head: t -> (h = t>>5, a = t&31); layernorm over 32, smooth leaky relu, dot
  const float* xa = Y0 + (size_t)e * 768 + 384;   // x_alpha = extra[0:256]
  float v = xa[t];
  red[t] = v; __syncthreads();
  for (int o = 16; o > 0; o >>= 1) { if ((t & 31) < o) red[t] += red[t + o]; __syncthreads(); }
  float mu = red[t & ~31] / 32.f; __syncthreads();
  float d = v - mu;
  red[t] = d * d; __syncthreads();
  for (int o = 16; o > 0; o >>= 1) { if ((t & 31) < o) red[t] += red[t + o]; __syncthreads(); }
  float var = red[t & ~31] / 32.f; __syncthreads();
  float ln = d * rsqrtf(var + 1e-5f) * alng[t & 31] + alnb[t & 31];
  float sg = 1.f / (1.f + expf(-ln));
  float sl = 0.6f * ln + 0.4f * ln * (2.f * sg - 1.f);  // smooth leaky relu a=0.2
  red[t] = sl * adot[t]; __syncthreads();
  for (int o = 16; o > 0; o >>= 1) { if ((t & 31) < o) red[t] += red[t + o]; __syncthreads(); }
  if ((t & 31) == 0) aout[(size_t)e * 8 + (t >> 5)] = red[t];
}

// ---------------------------------------------------------------------------
// K8: segment softmax over edge_dst
// ---------------------------------------------------------------------------
__global__ void amax_kernel(const float* __restrict__ a, const int* __restrict__ edst,
                            unsigned* __restrict__ amaxk) {
  int i = blockIdx.x * 256 + threadIdx.x;
  if (i >= EE * 8) return;
  int e = i >> 3, h = i & 7;
  atomicMax(&amaxk[edst[e] * 8 + h], enc_f(a[i]));
}
__global__ void aexp_kernel(float* __restrict__ a, const int* __restrict__ edst,
                            const unsigned* __restrict__ amaxk, float* __restrict__ asum) {
  int i = blockIdx.x * 256 + threadIdx.x;
  if (i >= EE * 8) return;
  int e = i >> 3, h = i & 7;
  int d = edst[e] * 8 + h;
  float ex = expf(a[i] - dec_f(amaxk[d]));
  a[i] = ex;
  atomicAdd(&asum[d], ex);
}

// ---------------------------------------------------------------------------
// K9: attention-weighted scatter into node accumulator. grid=EE, block=256
// ---------------------------------------------------------------------------
__global__ void scatter_kernel(const unsigned short* __restrict__ valrot,
                               const float* __restrict__ aexp, const float* __restrict__ asum,
                               const int* __restrict__ edst, float* __restrict__ node) {
  int e = blockIdx.x, t = threadIdx.x;
  int dst = edst[e];
  for (int i = t; i < ROW; i += 256) {
    int c = i & 127, h = c >> 4;                       // head = c / V, V=16
    float attn = aexp[(size_t)e * 8 + h] / asum[dst * 8 + h];
    atomicAdd(&node[(size_t)dst * ROW + i], attn * bf2f(valrot[(size_t)e * ROW + i]));
  }
}

// ---------------------------------------------------------------------------
// K10: x1 = x + so3_linear(node, proj_w, proj_b). grid=NN, block=128
// ---------------------------------------------------------------------------
__global__ void proj_kernel(const float* __restrict__ x, const float* __restrict__ node,
                            const float* __restrict__ pw, const float* __restrict__ pb,
                            float* __restrict__ x1) {
  int n = blockIdx.x, c = threadIdx.x;
  __shared__ float row[128];
  const int LOF[SS] = {0, 1, 1, 1, 2, 2, 2, 2, 2};
  for (int s = 0; s < SS; ++s) {
    row[c] = node[(size_t)n * ROW + s * CC + c];
    __syncthreads();
    int l = LOF[s];
    float acc = (s == 0) ? pb[c] : 0.f;
    for (int k = 0; k < 128; ++k) acc += row[k] * pw[l * 128 * 128 + k * 128 + c];
    x1[(size_t)n * ROW + s * CC + c] = x[(size_t)n * ROW + s * CC + c] + acc;
    __syncthreads();
  }
}

// ---------------------------------------------------------------------------
// K12: FFN branch + residual -> output. grid=NN, block=128
// ---------------------------------------------------------------------------
__global__ void ffn_kernel(const float* __restrict__ x1, const float* __restrict__ xn2,
                           const float* __restrict__ gw, const float* __restrict__ gb,
                           const float* __restrict__ fw1, const float* __restrict__ fb1,
                           const float* __restrict__ fw2, const float* __restrict__ fb2,
                           const float* __restrict__ togrid, const float* __restrict__ fromgrid,
                           float* __restrict__ out) {
  int n = blockIdx.x, c = threadIdx.x;
  __shared__ float xs[SS][CC];
  __shared__ float hh[SS][CC];
  __shared__ float tg[162], fg[162];
  for (int i = c; i < 162; i += 128) { tg[i] = togrid[i]; fg[i] = fromgrid[i]; }
  for (int s = 0; s < SS; ++s) xs[s][c] = xn2[(size_t)n * ROW + s * CC + c];
  __syncthreads();
  const int LOF[SS] = {0, 1, 1, 1, 2, 2, 2, 2, 2};
  float g = gb[c];
  for (int k = 0; k < 128; ++k) g += xs[0][k] * gw[k * 128 + c];
  float hl[SS];
#pragma unroll
  for (int s = 0; s < SS; ++s) {
    int l = LOF[s];
    float acc = (s == 0) ? fb1[c] : 0.f;
    for (int k = 0; k < 128; ++k) acc += xs[s][k] * fw1[l * 16384 + k * 128 + c];
    hl[s] = acc;
  }
  float tacc[SS];
#pragma unroll
  for (int s = 0; s < SS; ++s) tacc[s] = 0.f;
  for (int gi = 0; gi < 18; ++gi) {
    float gv = 0.f;
#pragma unroll
    for (int s = 0; s < SS; ++s) gv += tg[gi * 9 + s] * hl[s];
    float sg = siluf(gv);
#pragma unroll
    for (int s = 0; s < SS; ++s) tacc[s] += fg[s * 18 + gi] * sg;
  }
  hl[0] = siluf(g);
#pragma unroll
  for (int s = 1; s < SS; ++s) hl[s] = tacc[s];
  for (int s = 0; s < SS; ++s) hh[s][c] = hl[s];
  __syncthreads();
#pragma unroll
  for (int s = 0; s < SS; ++s) {
    int l = LOF[s];
    float acc = (s == 0) ? fb2[c] : 0.f;
    for (int k = 0; k < 128; ++k) acc += hh[s][k] * fw2[l * 16384 + k * 128 + c];
    out[(size_t)n * ROW + s * CC + c] = x1[(size_t)n * ROW + s * CC + c] + acc;
  }
}

// ---------------------------------------------------------------------------
// Host: orchestrate full forward pass
// ---------------------------------------------------------------------------
static size_t alloc_off(size_t& off, size_t bytes) {
  size_t r = off;
  off += (bytes + 255) & ~(size_t)255;
  return r;
}

extern "C" void kernel_launch(void* const* d_in, const int* in_sizes, int n_in,
                              void* d_out, int out_size, void* d_ws, size_t ws_size,
                              hipStream_t stream) {
  (void)in_sizes; (void)n_in; (void)out_size; (void)ws_size;
  const float* x        = (const float*)d_in[0];
  const float* edist    = (const float*)d_in[1];
  const float* wigner   = (const float*)d_in[2];
  const float* rad_w1   = (const float*)d_in[3];
  const float* rad_b1   = (const float*)d_in[4];
  const float* rad_ln_g = (const float*)d_in[5];
  const float* rad_ln_b = (const float*)d_in[6];
  const float* rad_w2   = (const float*)d_in[7];
  const float* rad_b2   = (const float*)d_in[8];
  const float* w1_m0    = (const float*)d_in[9];
  const float* b1_m0    = (const float*)d_in[10];
  const float* w1_m1    = (const float*)d_in[11];
  const float* w1_m2    = (const float*)d_in[12];
  const float* w2_m0    = (const float*)d_in[13];
  const float* b2_m0    = (const float*)d_in[14];
  const float* w2_m1    = (const float*)d_in[15];
  const float* w2_m2    = (const float*)d_in[16];
  const float* aln_g    = (const float*)d_in[17];
  const float* aln_b    = (const float*)d_in[18];
  const float* adot     = (const float*)d_in[19];
  const float* proj_w   = (const float*)d_in[20];
  const float* proj_b   = (const float*)d_in[21];
  const float* norm1_g  = (const float*)d_in[22];
  const float* norm1_b  = (const float*)d_in[23];
  const float* norm2_g  = (const float*)d_in[24];
  const float* norm2_b  = (const float*)d_in[25];
  const float* gate_w   = (const float*)d_in[26];
  const float* gate_b   = (const float*)d_in[27];
  const float* ffn_w1   = (const float*)d_in[28];
  const float* ffn_b1   = (const float*)d_in[29];
  const float* ffn_w2   = (const float*)d_in[30];
  const float* ffn_b2   = (const float*)d_in[31];
  const float* togrid   = (const float*)d_in[32];
  const float* fromgrid = (const float*)d_in[33];
  const int*   esrc     = (const int*)d_in[34];
  const int*   edst     = (const int*)d_in[35];
  float* out = (float*)d_out;

  char* ws = (char*)d_ws;
  size_t off = 0;
  unsigned short* w1m0T = (unsigned short*)(ws + alloc_off(off, (size_t)768*768*2));
  unsigned short* w1m1T = (unsigned short*)(ws + alloc_off(off, (size_t)512*512*2));
  unsigned short* w1m2T = (unsigned short*)(ws + alloc_off(off, (size_t)256*256*2));
  unsigned short* w2m0T = (unsigned short*)(ws + alloc_off(off, (size_t)384*384*2));
  unsigned short* w2m1T = (unsigned short*)(ws + alloc_off(off, (size_t)512*256*2));
  unsigned short* w2m2T = (unsigned short*)(ws + alloc_off(off, (size_t)256*128*2));
  float* xn   = (float*)(ws + alloc_off(off, (size_t)NN*ROW*4));      // later: x1
  float* rad  = (float*)(ws + alloc_off(off, (size_t)EE*768*4));      // later: Z0
  unsigned short* msgm = (unsigned short*)(ws + alloc_off(off, (size_t)EE*2304*2)); // later: valrot
  float* Y0   = (float*)(ws + alloc_off(off, (size_t)EE*768*4));
  float* YR1  = (float*)(ws + alloc_off(off, (size_t)EE*512*4));      // later: ZR1
  float* YI1  = (float*)(ws + alloc_off(off, (size_t)EE*512*4));      // later: ZI1
  float* YR2  = (float*)(ws + alloc_off(off, (size_t)EE*256*4));      // later: ZR2
  float* YI2  = (float*)(ws + alloc_off(off, (size_t)EE*256*4));      // later: ZI2
  unsigned short* hid2m = (unsigned short*)(ws + alloc_off(off, (size_t)EE*ROW*2));
  float*    aArr  = (float*)(ws + alloc_off(off, (size_t)EE*8*4));
  unsigned* amaxk = (unsigned*)(ws + alloc_off(off, (size_t)NN*8*4));
  float*    asum  = (float*)(ws + alloc_off(off, (size_t)NN*8*4));
  float* node = (float*)(ws + alloc_off(off, (size_t)NN*ROW*4));
  float* xn2  = (float*)(ws + alloc_off(off, (size_t)NN*ROW*4));
  float* x1 = xn;               // aliases (lifetimes disjoint, stream-ordered)
  float* Z0 = rad;
  float* ZR1 = YR1; float* ZI1 = YI1; float* ZR2 = YR2; float* ZI2 = YI2;
  unsigned short* valrot = msgm;

  // accumulator / softmax state init (required every call)
  hipMemsetAsync(amaxk, 0, (size_t)NN*8*4, stream);
  hipMemsetAsync(asum,  0, (size_t)NN*8*4, stream);
  hipMemsetAsync(node,  0, (size_t)NN*ROW*4, stream);

  // weight transposes to bf16 [N,K]
  auto cvt = [&](const float* w, unsigned short* o, int K, int N) {
    convT_bf16<<<(K*N + 255)/256, 256, 0, stream>>>(w, o, K, N);
  };
  cvt(w1_m0, w1m0T, 768, 768);
  cvt(w1_m1, w1m1T, 512, 512);
  cvt(w1_m2, w1m2T, 256, 256);
  cvt(w2_m0, w2m0T, 384, 384);
  cvt(w2_m1, w2m1T, 256, 512);
  cvt(w2_m2, w2m2T, 128, 256);

  eqv_norm_kernel<<<NN, 128, 0, stream>>>(x, norm1_g, norm1_b, xn);
  radial_kernel<<<EE, 256, 0, stream>>>(edist, rad_w1, rad_b1, rad_ln_g, rad_ln_b,
                                        rad_w2, rad_b2, rad);
  build_msg_kernel<<<EE, 256, 0, stream>>>(xn, wigner, rad, esrc, edst, msgm);

  // SO(2) conv 1 (WMMA + TDM): block tile 128x64
  dim3 gy((unsigned)1, (unsigned)((EE + 127) / 128));
  gy.x = 768/64; gemm_bf16_wmma<<<gy, 256, 0, stream>>>(msgm + 0,    2304, w1m0T, b1_m0,  Y0,  EE, 768, 768);
  gy.x = 512/64; gemm_bf16_wmma<<<gy, 256, 0, stream>>>(msgm + 768,  2304, w1m1T, nullptr, YR1, EE, 512, 512);
  gy.x = 512/64; gemm_bf16_wmma<<<gy, 256, 0, stream>>>(msgm + 1280, 2304, w1m1T, nullptr, YI1, EE, 512, 512);
  gy.x = 256/64; gemm_bf16_wmma<<<gy, 256, 0, stream>>>(msgm + 1792, 2304, w1m2T, nullptr, YR2, EE, 256, 256);
  gy.x = 256/64; gemm_bf16_wmma<<<gy, 256, 0, stream>>>(msgm + 2048, 2304, w1m2T, nullptr, YI2, EE, 256, 256);

  mid_kernel<<<EE, 128, 0, stream>>>(Y0, YR1, YI1, YR2, YI2, togrid, fromgrid, hid2m);

  // SO(2) conv 2 (WMMA + TDM)
  gy.x = 384/64; gemm_bf16_wmma<<<gy, 256, 0, stream>>>(hid2m + 0,    ROW, w2m0T, b2_m0,  Z0,  EE, 384, 384);
  gy.x = 512/64; gemm_bf16_wmma<<<gy, 256, 0, stream>>>(hid2m + 384,  ROW, w2m1T, nullptr, ZR1, EE, 512, 256);
  gy.x = 512/64; gemm_bf16_wmma<<<gy, 256, 0, stream>>>(hid2m + 640,  ROW, w2m1T, nullptr, ZI1, EE, 512, 256);
  gy.x = 256/64; gemm_bf16_wmma<<<gy, 256, 0, stream>>>(hid2m + 896,  ROW, w2m2T, nullptr, ZR2, EE, 256, 128);
  gy.x = 256/64; gemm_bf16_wmma<<<gy, 256, 0, stream>>>(hid2m + 1024, ROW, w2m2T, nullptr, ZI2, EE, 256, 128);

  post_kernel<<<EE, 256, 0, stream>>>(Z0, ZR1, ZI1, ZR2, ZI2, Y0, wigner,
                                      aln_g, aln_b, adot, valrot, aArr);

  amax_kernel<<<(EE*8 + 255)/256, 256, 0, stream>>>(aArr, edst, amaxk);
  aexp_kernel<<<(EE*8 + 255)/256, 256, 0, stream>>>(aArr, edst, amaxk, asum);
  scatter_kernel<<<EE, 256, 0, stream>>>(valrot, aArr, asum, edst, node);

  proj_kernel<<<NN, 128, 0, stream>>>(x, node, proj_w, proj_b, x1);
  eqv_norm_kernel<<<NN, 128, 0, stream>>>(x1, norm2_g, norm2_b, xn2);
  ffn_kernel<<<NN, 128, 0, stream>>>(x1, xn2, gate_w, gate_b, ffn_w1, ffn_b1,
                                     ffn_w2, ffn_b2, togrid, fromgrid, out);
}
